// uOT_Loss_52123723104492
// MI455X (gfx1250) — matrix operand order repaired and checked by reading, not compiled, not run on hardware
//
#include <hip/hip_runtime.h>

// ---------------- problem constants (from reference) ----------------
#define IMSIZE 256
#define Gdim   128
#define NROWS  16384          // G*G grid cells
#define MCOLS  256            // points per sample
#define BSAMP  4
#define NITER  50
#define INV_EPS 10.0f         // 1/EPS, EPS = 0.1
#define NTE    (-0.0909090909090909f)   // -tau*eps, tau = RHO/(RHO+EPS) = 1/1.1
#define LAMB   0.00535428f    // 0.8 * sigmoid(10*(0/400 - 0.5)) = 0.8*sigmoid(-5)
#define WSCALE 1.05f          // RHO + EPS/2

typedef __attribute__((ext_vector_type(16))) __bf16       v16bf;
typedef __attribute__((ext_vector_type(8)))  float        v8f;
typedef __attribute__((ext_vector_type(8)))  unsigned int v8u;

union ABfrag { v8u u; v16bf h; };
struct Atile { uint4 lo, hi; };

// round-to-nearest-even float -> bf16 bits
__device__ inline unsigned short f2bf(float x) {
    unsigned int b = __float_as_uint(x);
    b = (b + 0x7FFFu + ((b >> 16) & 1u)) >> 16;
    return (unsigned short)b;
}
__device__ inline unsigned int pack_bf16(float x, float y) {
    return (unsigned int)f2bf(x) | ((unsigned int)f2bf(y) << 16);
}

// A-fragment (16x32 bf16 tile) load, in contiguous k order (same as load_B):
// lanes 0-15: row=lane, K = 0..15; lanes 16-31: row=lane-16, K = 16..31
__device__ inline Atile load_A(const unsigned short* __restrict__ Kbase, int ldcols,
                               int rbase, int c0) {
    const int lane = threadIdx.x & 31;
    const int row  = rbase + (lane & 15);
    const int hiA  = (lane < 16) ? 0 : 16;
    const char* rp = (const char*)(Kbase + (size_t)row * (size_t)ldcols + c0 + hiA);
    Atile t;
    t.lo = *(const uint4*)(rp);        // elements 0..7
    t.hi = *(const uint4*)(rp + 16);   // elements 8..15 (K offset +8)
    return t;
}

// B-fragment: broadcast vector chunk v[c0 .. c0+31] replicated across 16 columns.
// lanes 0-15: K=0..15; lanes 16-31: K=16..31 (contiguous k order, same as load_A)
__device__ inline ABfrag load_B(const unsigned int* vlds, int c0) {
    const int lane = threadIdx.x & 31;
    const int hiB  = (lane < 16) ? 0 : 16;
    const uint4* vp = (const uint4*)(vlds + ((c0 + hiB) >> 1));
    ABfrag bm;
    uint4 b0 = vp[0], b1 = vp[1];
    bm.u[0] = b0.x; bm.u[1] = b0.y; bm.u[2] = b0.z; bm.u[3] = b0.w;
    bm.u[4] = b1.x; bm.u[5] = b1.y; bm.u[6] = b1.z; bm.u[7] = b1.w;
    return bm;
}

__device__ inline v8f wmma_mac(const Atile& t, const ABfrag& bm, v8f c) {
    ABfrag a;
    a.u[0] = t.lo.x; a.u[1] = t.lo.y; a.u[2] = t.lo.z; a.u[3] = t.lo.w;
    a.u[4] = t.hi.x; a.u[5] = t.hi.y; a.u[6] = t.hi.z; a.u[7] = t.hi.w;
    return __builtin_amdgcn_wmma_f32_16x16x32_bf16(false, a.h, false, bm.h,
                                                   (short)0, c, false, false);
}

// ---------------- K build: K[b][i][j] = exp(-5*max(d2,0)) in bf16 ----------------
__global__ void build_K(const float* __restrict__ grid, const float* __restrict__ pts,
                        unsigned short* __restrict__ K) {
    const int b = blockIdx.y;
    const int j = threadIdx.x;                         // 0..255 = column
    const float yx = pts[((size_t)b * MCOLS + j) * 2 + 0] * (1.0f / IMSIZE);
    const float yy = pts[((size_t)b * MCOLS + j) * 2 + 1] * (1.0f / IMSIZE);
    const float yn = yx * yx + yy * yy;
    unsigned short* Kb = K + (size_t)b * NROWS * MCOLS;
    const int i0 = blockIdx.x * 4;
    for (int r = 0; r < 4; ++r) {
        const int i = i0 + r;
        const float xx = grid[i * 2 + 0], xy = grid[i * 2 + 1];
        float d2 = xx * xx + xy * xy + yn - 2.0f * (xx * yx + xy * yy);
        d2 = fmaxf(d2, 0.0f);
        Kb[(size_t)i * MCOLS + j] = f2bf(__expf(-5.0f * d2));   // exp(-0.5*d2/eps)
    }
}

// KT[b][j][i] (row-major M x N), coalesced along i
__global__ void build_KT(const float* __restrict__ grid, const float* __restrict__ pts,
                         unsigned short* __restrict__ KT) {
    const int b = blockIdx.y;
    const int j = blockIdx.x;
    const float yx = pts[((size_t)b * MCOLS + j) * 2 + 0] * (1.0f / IMSIZE);
    const float yy = pts[((size_t)b * MCOLS + j) * 2 + 1] * (1.0f / IMSIZE);
    const float yn = yx * yx + yy * yy;
    unsigned short* row = KT + ((size_t)b * MCOLS + j) * NROWS;
    for (int i = threadIdx.x; i < NROWS; i += blockDim.x) {
        const float xx = grid[i * 2 + 0], xy = grid[i * 2 + 1];
        float d2 = xx * xx + xy * xy + yn - 2.0f * (xx * yx + xy * yy);
        d2 = fmaxf(d2, 0.0f);
        row[i] = f2bf(__expf(-5.0f * d2));
    }
}

__global__ void init_misc(const float* __restrict__ dens, float* __restrict__ a,
                          float* __restrict__ bv1, float* __restrict__ g) {
    const int idx = blockIdx.x * blockDim.x + threadIdx.x;
    if (idx < BSAMP * NROWS) a[idx] = fmaxf(dens[idx], 1e-8f);
    if (idx < BSAMP * MCOLS) { bv1[idx] = 1.0f; g[idx] = 0.0f; }
}

// f-sweep: out_i = g(s_i), s_i = sum_j wt_j*exp(pot_j/eps)*K[b][i][j]
// grid (NROWS/512, BSAMP), block 256 (8 waves; each wave owns its 16-row tiles)
__global__ void f_update(const unsigned short* __restrict__ Kws,
                         const float* __restrict__ pot, const float* __restrict__ wt,
                         float* __restrict__ out, int mode) {
    __shared__ __align__(16) unsigned int vlds[MCOLS / 2];
    const int b = blockIdx.y;
    const float* potb = pot + (size_t)b * MCOLS;
    const float* wtb  = wt  + (size_t)b * MCOLS;
    if (threadIdx.x < MCOLS / 2) {
        const int j = threadIdx.x * 2;
        vlds[threadIdx.x] = pack_bf16(wtb[j]     * __expf(potb[j]     * INV_EPS),
                                      wtb[j + 1] * __expf(potb[j + 1] * INV_EPS));
    }
    __syncthreads();
    // entire broadcast vector (8 B-fragments) lives in registers, loop-invariant
    ABfrag bfr[MCOLS / 32];
#pragma unroll
    for (int ch = 0; ch < MCOLS / 32; ++ch) bfr[ch] = load_B(vlds, ch * 32);

    const unsigned short* Kb = Kws + (size_t)b * NROWS * MCOLS;
    const int wave = threadIdx.x >> 5, lane = threadIdx.x & 31;
    float* outb = out + (size_t)b * NROWS;
    for (int r = 0; r < 4; ++r) {
        const int rbase = (blockIdx.x * 32 + r * 8 + wave) * 16;
        // issue ALL tile loads first (one clause), then the straight WMMA chain
        Atile at[MCOLS / 32];
#pragma unroll
        for (int ch = 0; ch < MCOLS / 32; ++ch) at[ch] = load_A(Kb, MCOLS, rbase, ch * 32);
        v8f c = {0.f, 0.f, 0.f, 0.f, 0.f, 0.f, 0.f, 0.f};
#pragma unroll
        for (int ch = 0; ch < MCOLS / 32; ++ch) c = wmma_mac(at[ch], bfr[ch], c);
        if ((lane & 15) == 0) {                      // col 0 of D: lanes 0 & 16
            const int ro = rbase + ((lane >> 4) << 3);
#pragma unroll
            for (int k = 0; k < 8; ++k) {
                const float s = c[k];
                outb[ro + k] = (mode == 0) ? NTE * __logf(s) : s;
            }
        }
    }
}

// g-sweep: out_j = g(t_j), t_j = sum_i wt_i*exp(pot_i/eps)*KT[b][j][i]
// grid (MCOLS/16, BSAMP), block 256: 8 waves split the 512 K-chunks of one 16-row tile
__global__ void g_update(const unsigned short* __restrict__ KTws,
                         const float* __restrict__ pot, const float* __restrict__ wt,
                         float* __restrict__ out, int mode) {
    __shared__ __align__(16) unsigned int vlds[NROWS / 2];   // 32 KB of 320 KB LDS
    __shared__ float partial[8][16];
    const int b = blockIdx.y;
    const float* potb = pot + (size_t)b * NROWS;
    const float* wtb  = wt  + (size_t)b * NROWS;
    for (int p = threadIdx.x; p < NROWS / 2; p += blockDim.x) {
        const int i = p * 2;
        vlds[p] = pack_bf16(wtb[i]     * __expf(potb[i]     * INV_EPS),
                            wtb[i + 1] * __expf(potb[i + 1] * INV_EPS));
    }
    __syncthreads();
    const unsigned short* Kb = KTws + (size_t)b * MCOLS * NROWS;
    const int wave = threadIdx.x >> 5, lane = threadIdx.x & 31;
    const int rbase = blockIdx.x * 16;

    // 2-deep software pipeline: next chunk's loads in flight during current WMMA.
    // Pipeline condition is wave-uniform -> EXEC stays all-ones across every WMMA.
    v8f c = {0.f, 0.f, 0.f, 0.f, 0.f, 0.f, 0.f, 0.f};
    Atile  acur = load_A(Kb, NROWS, rbase, wave * 32);
    ABfrag bcur = load_B(vlds, wave * 32);
    for (int ch = wave; ch < NROWS / 32; ch += 8) {
        Atile  anxt = acur;
        ABfrag bnxt = bcur;
        const int ch2 = ch + 8;
        if (ch2 < NROWS / 32) {
            anxt = load_A(Kb, NROWS, rbase, ch2 * 32);
            bnxt = load_B(vlds, ch2 * 32);
        }
        c = wmma_mac(acur, bcur, c);
        acur = anxt;
        bcur = bnxt;
    }
    if ((lane & 15) == 0) {
        const int rr = (lane >> 4) << 3;
#pragma unroll
        for (int k = 0; k < 8; ++k) partial[wave][rr + k] = c[k];
    }
    __syncthreads();
    if (threadIdx.x < 16) {
        float s = 0.f;
#pragma unroll
        for (int w = 0; w < 8; ++w) s += partial[w][threadIdx.x];
        out[(size_t)b * MCOLS + rbase + threadIdx.x] = (mode == 0) ? NTE * __logf(s) : s;
    }
}

// new_b_j = clip(1 + lamb*(exp(g_j/eps)*t'_j - 1), 1e-8); reset g for run 2
__global__ void make_newb(float* __restrict__ g, const float* __restrict__ tp,
                          float* __restrict__ nb) {
    const int idx = blockIdx.x * blockDim.x + threadIdx.x;
    if (idx < BSAMP * MCOLS) {
        const float resid = __expf(g[idx] * INV_EPS) * tp[idx] - 1.0f;
        nb[idx] = fmaxf(1.0f + LAMB * resid, 1e-8f);
        g[idx] = 0.0f;
    }
}

// loss = 1.05 * sum_b [ sum_i a_i(1-e^{-F2_i}) + sum_j nb_j(1-e^{-G2_j}) ]
__global__ void loss_reduce(const float* __restrict__ a, const float* __restrict__ F2,
                            const float* __restrict__ nb, const float* __restrict__ G2,
                            float* __restrict__ outv) {
    __shared__ float red[256];
    float acc = 0.f;
    for (int i = threadIdx.x; i < BSAMP * NROWS; i += 256)
        acc += a[i] * (1.0f - __expf(-F2[i]));
    for (int j = threadIdx.x; j < BSAMP * MCOLS; j += 256)
        acc += nb[j] * (1.0f - __expf(-G2[j]));
    red[threadIdx.x] = acc;
    __syncthreads();
    for (int s = 128; s > 0; s >>= 1) {
        if (threadIdx.x < s) red[threadIdx.x] += red[threadIdx.x + s];
        __syncthreads();
    }
    if (threadIdx.x == 0) outv[0] = WSCALE * red[0];
}

extern "C" void kernel_launch(void* const* d_in, const int* in_sizes, int n_in,
                              void* d_out, int out_size, void* d_ws, size_t ws_size,
                              hipStream_t stream) {
    const float* dens = (const float*)d_in[0];   // (B,1,G,G) fp32
    const float* pts  = (const float*)d_in[1];   // (B,M,2) fp32
    const float* grid = (const float*)d_in[2];   // (N,2) fp32

    char* ws = (char*)d_ws;
    const size_t SZ_K = (size_t)BSAMP * NROWS * MCOLS * sizeof(unsigned short); // 32 MB
    unsigned short* K  = (unsigned short*)(ws);
    unsigned short* KT = (unsigned short*)(ws + SZ_K);
    float* a   = (float*)(ws + 2 * SZ_K);        // both K copies resident in 192MB L2
    float* f   = a   + BSAMP * NROWS;
    float* g   = f   + BSAMP * NROWS;
    float* bv1 = g   + BSAMP * MCOLS;
    float* nb  = bv1 + BSAMP * MCOLS;
    float* tp  = nb  + BSAMP * MCOLS;
    (void)in_sizes; (void)n_in; (void)out_size; (void)ws_size;

    build_K <<<dim3(NROWS / 4, BSAMP), 256, 0, stream>>>(grid, pts, K);
    build_KT<<<dim3(MCOLS,     BSAMP), 256, 0, stream>>>(grid, pts, KT);
    init_misc<<<(BSAMP * NROWS + 255) / 256, 256, 0, stream>>>(dens, a, bv1, g);

    // ---- run 1: Sinkhorn with b = 1 (recompute_beta) ----
    for (int it = 0; it < NITER; ++it) {
        f_update<<<dim3(32, BSAMP), 256, 0, stream>>>(K,  g, bv1, f, 0);
        g_update<<<dim3(16, BSAMP), 256, 0, stream>>>(KT, f, a,   g, 0);
    }
    f_update<<<dim3(32, BSAMP), 256, 0, stream>>>(K,  g, bv1, f, 0);  // F = f51
    g_update<<<dim3(16, BSAMP), 256, 0, stream>>>(KT, f, a,  tp, 1);  // t' raw sums
    make_newb<<<(BSAMP * MCOLS + 255) / 256, 256, 0, stream>>>(g, tp, nb);

    // ---- run 2: Sinkhorn with new_b (loss pass) ----
    for (int it = 0; it < NITER; ++it) {
        f_update<<<dim3(32, BSAMP), 256, 0, stream>>>(K,  g, nb, f, 0);
        g_update<<<dim3(16, BSAMP), 256, 0, stream>>>(KT, f, a,  g, 0);
    }
    f_update<<<dim3(32, BSAMP), 256, 0, stream>>>(K, g, nb, f, 0);    // F2 = f51
    loss_reduce<<<1, 256, 0, stream>>>(a, f, nb, g, (float*)d_out);   // G2 = current g
}
